// Net_35055523070559
// MI455X (gfx1250) — matrix operand order repaired and checked by reading, not compiled, not run on hardware
//
#include <hip/hip_runtime.h>
#include <hip/hip_bf16.h>

#define BB 1024
#define PP 128
#define KN 20
#define EE (BB * PP * KN)      // 2,621,440 edges
#define BP (BB * PP)           // 131,072 points
#define OUTC 2
#define STAT_BLOCKS 1280       // EE / (256*8)

typedef __attribute__((ext_vector_type(16))) _Float16 v16h;
typedef __attribute__((ext_vector_type(8)))  float    v8f;
typedef __attribute__((ext_vector_type(2)))  float    v2f;
typedef __attribute__((ext_vector_type(4)))  unsigned v4u;
typedef __attribute__((ext_vector_type(8)))  int      v8i;
typedef __attribute__((ext_vector_type(4)))  int      v4i;

#if defined(__has_builtin)
#if __has_builtin(__builtin_amdgcn_wmma_f32_16x16x4_f32)
#define HAVE_WMMA_F32 1
#endif
#if __has_builtin(__builtin_amdgcn_tensor_load_to_lds)
#define HAVE_TDM 1
#endif
#endif

// ---- compile-only probes (visible in stderr) --------------------------
#if defined(HAVE_WMMA_F32)
#warning "PROBE: f32 16x16x4 WMMA path ENABLED"
#else
#warning "PROBE: f32 16x16x4 WMMA path DISABLED (scalar fallback)"
#endif
#if defined(HAVE_TDM)
#if __has_include(<hip/amd_detail/amd_gfx1250_TDM.h>)
#warning "PROBE: TDM path ENABLED (6-arg builtin)"
#else
#warning "PROBE: TDM path ENABLED (5-arg builtin)"
#endif
#else
#warning "PROBE: TDM path DISABLED (plain-load fallback)"
#endif

static __device__ __forceinline__ v8f wmma16(v16h a, v16h b, v8f c) {
  return __builtin_amdgcn_wmma_f32_16x16x32_f16(
      /*neg_a=*/false, a, /*neg_b=*/false, b,
      /*c_mod=*/(short)0, c, /*reuse_a=*/false, /*reuse_b=*/false);
}

// =======================================================================
// kNN via WMMA Gram matrix: d2 = n_i + n_j - 2 * X X^T  (fp32, exact)
// block = 128 thr (4 waves) = one graph; two 64-row halves of the 128x128
// distance matrix to bound LDS.  NC = real feature dim, FP = padded (%4).
// =======================================================================
template <int NC, int FP>
__global__ __launch_bounds__(128) void knn_wmma_kernel(const float* __restrict__ x,
                                                       int* __restrict__ idx) {
  __shared__ float xs[PP][FP];
  __shared__ float nrm[PP];
  __shared__ float d2[PP / 2][PP];
  const int tid = threadIdx.x, lane = tid & 31, wave = tid >> 5;
  const int r16 = lane & 15, half = lane >> 4;
  const int b = blockIdx.x;

  // ---- stage features into LDS (TDM for the dense un-padded case) ------
#if defined(HAVE_TDM)
  if constexpr (NC == FP) {
    if (wave == 0) {
      const unsigned lds_base = (unsigned)(uintptr_t)&xs[0][0];
      const unsigned long long ga =
          (unsigned long long)(uintptr_t)(x + (size_t)b * PP * NC);
      const unsigned elems = PP * NC;             // 2048 x 4B, contiguous
      v4u g0;
      g0[0] = 1u;                                  // count=1, user mode
      g0[1] = lds_base;                            // lds_addr
      g0[2] = (unsigned)(ga & 0xFFFFFFFFu);        // global_addr[31:0]
      g0[3] = (unsigned)((ga >> 32) & 0x1FFFFFFu)  // global_addr[56:32]
              | (2u << 30);                        // type = 2 (image)
      v8i g1;
      g1[0] = (int)(2u << 16);                     // data_size = 4B
      g1[1] = (int)((elems & 0xFFFFu) << 16);      // tensor_dim0[15:0]
      g1[2] = (int)(((elems >> 16) & 0xFFFFu)      // tensor_dim0[31:16]
              | (1u << 16));                       // tensor_dim1 = 1
      g1[3] = (int)((elems & 0xFFFFu) << 16);      // tile_dim0 = 2048
      g1[4] = 1;                                   // tile_dim1 = 1
      g1[5] = (int)elems;                          // tensor_dim0_stride
      g1[6] = 0;
      g1[7] = 0;
      v4i gz; gz[0] = gz[1] = gz[2] = gz[3] = 0;
#if __has_include(<hip/amd_detail/amd_gfx1250_TDM.h>)
      v8i gz8; for (int i = 0; i < 8; ++i) gz8[i] = 0;
      __builtin_amdgcn_tensor_load_to_lds(g0, g1, gz, gz, gz8, 0);
#else
      __builtin_amdgcn_tensor_load_to_lds(g0, g1, gz, gz, 0);
#endif
      __builtin_amdgcn_s_wait_tensorcnt(0);
    }
    __syncthreads();
    float s = 0.0f;
    for (int f = 0; f < NC; ++f) s += xs[tid][f] * xs[tid][f];
    nrm[tid] = s;
    __syncthreads();
  } else
#endif
  {
    for (int f = 0; f < FP; ++f)
      xs[tid][f] = (f < NC) ? x[((size_t)b * PP + tid) * NC + f] : 0.0f;
    float s = 0.0f;
    for (int f = 0; f < NC; ++f) s += xs[tid][f] * xs[tid][f];
    nrm[tid] = s;
    __syncthreads();
  }

  // ---- two 64-row halves ----------------------------------------------
  for (int h = 0; h < 2; ++h) {
    const int mi = h * 4 + wave;                  // wave owns one M row-block
    for (int nj = 0; nj < 8; ++nj) {
      const int arow = mi * 16 + r16;
      const int brow = nj * 16 + r16;
      v8f acc; for (int r = 0; r < 8; ++r) acc[r] = 0.0f;
#if defined(HAVE_WMMA_F32)
      for (int s = 0; s < FP / 4; ++s) {
        const int k0 = s * 4 + half * 2;          // A16x4: lanes<16 K={0,1}
        v2f a, bv;
        a[0]  = xs[arow][k0]; a[1]  = xs[arow][k0 + 1];
        bv[0] = xs[brow][k0]; bv[1] = xs[brow][k0 + 1];
        acc = __builtin_amdgcn_wmma_f32_16x16x4_f32(
            false, a, false, bv, (short)0, acc, false, false);
      }
#else
      for (int r = 0; r < 8; ++r) {
        const int row = mi * 16 + r + half * 8;
        float s = 0.0f;
        for (int f = 0; f < FP; ++f) s += xs[row][f] * xs[brow][f];
        acc[r] = s;
      }
#endif
      for (int r = 0; r < 8; ++r) {
        const int row  = mi * 16 + r + half * 8;
        const int colp = nj * 16 + r16;
        float v = nrm[row] + nrm[colp] - 2.0f * acc[r];
        d2[row - h * 64][colp] = (row == colp) ? 3.0e38f : v;
      }
    }
    __syncthreads();
    if (tid < 64) {
      const int i = h * 64 + tid;
      float bd[KN]; int bi_[KN];
      for (int k = 0; k < KN; ++k) { bd[k] = 3.0e38f; bi_[k] = 0; }
      for (int j = 0; j < PP; ++j) {
        const float d = d2[tid][j];               // diagonal = +inf sentinel
        if (d < bd[KN - 1]) {
          int p = KN - 1;
          while (p > 0 && bd[p - 1] > d) { bd[p] = bd[p - 1]; bi_[p] = bi_[p - 1]; --p; }
          bd[p] = d; bi_[p] = j;
        }
      }
      for (int k = 0; k < KN; ++k) idx[((size_t)b * PP + i) * KN + k] = bi_[k];
    }
    __syncthreads();
  }
}

// -------------------------------------------------- BN statistics, layer 1
__global__ __launch_bounds__(256) void stats1_kernel(const float* __restrict__ pos,
                                                     const int* __restrict__ idx,
                                                     const float* __restrict__ w1,
                                                     const float* __restrict__ b1,
                                                     float* __restrict__ partial) {
  __shared__ float wr[8][32];
  const int tid = threadIdx.x, lane = tid & 31, wave = tid >> 5;
  float s[16], q[16];
  for (int c = 0; c < 16; ++c) { s[c] = 0.0f; q[c] = 0.0f; }
  for (int e = 0; e < 8; ++e) {
    int eg = (blockIdx.x * 256 + tid) * 8 + e;
    int gp = eg / KN, kk = eg % KN, b = gp >> 7;
    int jn = idx[gp * KN + kk];
    const float* pi = pos + (size_t)gp * 3;
    const float* pj = pos + (size_t)(b * PP + jn) * 3;
    float t[6];
    t[0] = pi[0]; t[1] = pi[1]; t[2] = pi[2];
    t[3] = pj[0] - pi[0]; t[4] = pj[1] - pi[1]; t[5] = pj[2] - pi[2];
    for (int c = 0; c < 16; ++c) {
      float v = b1[c];
      for (int f = 0; f < 6; ++f) v += t[f] * w1[f * 16 + c];
      s[c] += v; q[c] += v * v;
    }
  }
  for (int off = 16; off; off >>= 1)
    for (int c = 0; c < 16; ++c) {
      s[c] += __shfl_xor(s[c], off, 32);
      q[c] += __shfl_xor(q[c], off, 32);
    }
  if (lane == 0)
    for (int c = 0; c < 16; ++c) { wr[wave][c] = s[c]; wr[wave][16 + c] = q[c]; }
  __syncthreads();
  if (tid < 32) {
    float acc = 0.0f;
    for (int w = 0; w < 8; ++w) acc += wr[w][tid];
    partial[blockIdx.x * 32 + tid] = acc;
  }
}

// -------------------------------------------------- BN statistics, layer 2
__global__ __launch_bounds__(256) void stats2_kernel(const float* __restrict__ pos,
                                                     const int* __restrict__ idx,
                                                     const float* __restrict__ w1,
                                                     const float* __restrict__ b1,
                                                     const float* __restrict__ bn1,
                                                     const float* __restrict__ w2,
                                                     const float* __restrict__ b2,
                                                     float* __restrict__ partial) {
  __shared__ float wr[8][32];
  const int tid = threadIdx.x, lane = tid & 31, wave = tid >> 5;
  float s[16], q[16];
  for (int c = 0; c < 16; ++c) { s[c] = 0.0f; q[c] = 0.0f; }
  for (int e = 0; e < 8; ++e) {
    int eg = (blockIdx.x * 256 + tid) * 8 + e;
    int gp = eg / KN, kk = eg % KN, b = gp >> 7;
    int jn = idx[gp * KN + kk];
    const float* pi = pos + (size_t)gp * 3;
    const float* pj = pos + (size_t)(b * PP + jn) * 3;
    float t[6];
    t[0] = pi[0]; t[1] = pi[1]; t[2] = pi[2];
    t[3] = pj[0] - pi[0]; t[4] = pj[1] - pi[1]; t[5] = pj[2] - pi[2];
    float a1[16];
    for (int c = 0; c < 16; ++c) {
      float v = b1[c];
      for (int f = 0; f < 6; ++f) v += t[f] * w1[f * 16 + c];
      a1[c] = fmaxf(v * bn1[c] + bn1[16 + c], 0.0f);
    }
    for (int c = 0; c < 16; ++c) {
      float v = b2[c];
      for (int k = 0; k < 16; ++k) v += a1[k] * w2[k * 16 + c];
      s[c] += v; q[c] += v * v;
    }
  }
  for (int off = 16; off; off >>= 1)
    for (int c = 0; c < 16; ++c) {
      s[c] += __shfl_xor(s[c], off, 32);
      q[c] += __shfl_xor(q[c], off, 32);
    }
  if (lane == 0)
    for (int c = 0; c < 16; ++c) { wr[wave][c] = s[c]; wr[wave][16 + c] = q[c]; }
  __syncthreads();
  if (tid < 32) {
    float acc = 0.0f;
    for (int w = 0; w < 8; ++w) acc += wr[w][tid];
    partial[blockIdx.x * 32 + tid] = acc;
  }
}

// ------------------------------------------ BN finalize: scale/shift per ch
__global__ void bn_finalize_kernel(const float* __restrict__ partial, int nb,
                                   const float* __restrict__ gamma,
                                   const float* __restrict__ beta,
                                   float* __restrict__ bn) {
  int c = threadIdx.x;
  if (c >= 16) return;
  float sum = 0.0f, sq = 0.0f;
  for (int i = 0; i < nb; ++i) {
    sum += partial[i * 32 + c];
    sq  += partial[i * 32 + 16 + c];
  }
  const float invE = 1.0f / (float)EE;
  float mean = sum * invE;
  float var  = sq * invE - mean * mean;
  float sc   = gamma[c] * rsqrtf(var + 1e-5f);
  bn[c]      = sc;
  bn[16 + c] = beta[c] - mean * sc;
}

// -------------------------------- EdgeConv1: 3 chained WMMAs + max over K
__global__ __launch_bounds__(128) void conv1_kernel(
    const float* __restrict__ pos, const int* __restrict__ idx,
    const float* __restrict__ w1, const float* __restrict__ b1,
    const float* __restrict__ w2, const float* __restrict__ b2,
    const float* __restrict__ w3, const float* __restrict__ b3,
    const float* __restrict__ bn1, const float* __restrict__ bn2,
    float* __restrict__ x1) {
  __shared__ _Float16 sb[4][16][16];
  __shared__ float hb[320][16];
  const int tid = threadIdx.x, lane = tid & 31, wave = tid >> 5;
  const int r16 = lane & 15, half = lane >> 4, kb = half * 8;

  v16h B1, B2, B3;
  for (int j = 0; j < 16; ++j) { B1[j] = (_Float16)0; B2[j] = (_Float16)0; B3[j] = (_Float16)0; }
  if (half == 0) {
    for (int j = 0; j < 6;  ++j) B1[j] = (_Float16)w1[j * 16 + r16];
    for (int j = 0; j < 16; ++j) B2[j] = (_Float16)w2[j * 16 + r16];
    for (int j = 0; j < 16; ++j) B3[j] = (_Float16)w3[j * 16 + r16];
  }
  const float bb1 = b1[r16], bb2 = b2[r16], bb3 = b3[r16];
  const float s1 = bn1[r16], o1 = bn1[16 + r16];
  const float s2 = bn2[r16], o2 = bn2[16 + r16];

  for (int tt = 0; tt < 5; ++tt) {
    const int tile = tt * 4 + wave;
    const int e = tile * 16 + r16;
    const int pt = e / KN, kk = e % KN;
    const int gp = blockIdx.x * 16 + pt, b = gp >> 7;
    const int jn = idx[gp * KN + kk];
    const float* pi = pos + (size_t)gp * 3;
    const float* pj = pos + (size_t)(b * PP + jn) * 3;

    v16h A; for (int j = 0; j < 16; ++j) A[j] = (_Float16)0;
    if (half == 0) {
      A[0] = (_Float16)pi[0]; A[1] = (_Float16)pi[1]; A[2] = (_Float16)pi[2];
      A[3] = (_Float16)(pj[0] - pi[0]);
      A[4] = (_Float16)(pj[1] - pi[1]);
      A[5] = (_Float16)(pj[2] - pi[2]);
    }
    v8f acc; for (int r = 0; r < 8; ++r) acc[r] = bb1;
    acc = wmma16(A, B1, acc);

    __syncthreads();                      // prior-iteration sb reads done
    for (int r = 0; r < 8; ++r) {
      float v = fmaxf(acc[r] * s1 + o1, 0.0f);
      sb[wave][r + half * 8][r16] = (_Float16)v;
    }
    __syncthreads();

    v16h A2;
    for (int j = 0; j < 16; ++j) A2[j] = (half == 0) ? sb[wave][r16][j] : (_Float16)0;
    v8f acc2; for (int r = 0; r < 8; ++r) acc2[r] = bb2;
    acc2 = wmma16(A2, B2, acc2);

    __syncthreads();
    for (int r = 0; r < 8; ++r) {
      float v = fmaxf(acc2[r] * s2 + o2, 0.0f);
      sb[wave][r + half * 8][r16] = (_Float16)v;
    }
    __syncthreads();

    v16h A3;
    for (int j = 0; j < 16; ++j) A3[j] = (half == 0) ? sb[wave][r16][j] : (_Float16)0;
    v8f acc3; for (int r = 0; r < 8; ++r) acc3[r] = bb3;
    acc3 = wmma16(A3, B3, acc3);

    for (int r = 0; r < 8; ++r) hb[tile * 16 + r + half * 8][r16] = acc3[r];
  }
  __syncthreads();
  for (int o = tid; o < 256; o += 128) {
    const int pt = o >> 4, ch = o & 15;
    float m = -3.0e38f;
    for (int k = 0; k < KN; ++k) m = fmaxf(m, hb[pt * KN + k][ch]);
    x1[(size_t)(blockIdx.x * 16 + pt) * 16 + ch] = m;
  }
}

// ---------------------------------- EdgeConv2: 32->32 WMMA + max over K
__global__ __launch_bounds__(128) void conv2_kernel(
    const float* __restrict__ x1, const int* __restrict__ idx,
    const float* __restrict__ w, const float* __restrict__ bias,
    float* __restrict__ x2) {
  __shared__ float hb[320][32];
  const int tid = threadIdx.x, lane = tid & 31, wave = tid >> 5;
  const int r16 = lane & 15, half = lane >> 4, kb = half * 8;

  v16h Bt0, Bt1;
  for (int j = 0; j < 16; ++j) {
    Bt0[j] = (_Float16)w[(j < 8 ? kb + j : 8 + kb + j) * 32 + r16];
    Bt1[j] = (_Float16)w[(j < 8 ? kb + j : 8 + kb + j) * 32 + 16 + r16];
  }
  const float bb0 = bias[r16], bb1 = bias[16 + r16];

  for (int tt = 0; tt < 5; ++tt) {
    const int tile = tt * 4 + wave;
    const int e = tile * 16 + r16;
    const int pt = e / KN, kk = e % KN;
    const int gp = blockIdx.x * 16 + pt, b = gp >> 7;
    const int jn = idx[gp * KN + kk];
    const float* xi = x1 + (size_t)gp * 16;
    const float* xj = x1 + (size_t)(b * PP + jn) * 16;

    v16h A;
    for (int j = 0; j < 8; ++j) {
      float fi = xi[kb + j], fj = xj[kb + j];
      A[j]     = (_Float16)fi;
      A[8 + j] = (_Float16)(fj - fi);
    }
    v8f a0; for (int r = 0; r < 8; ++r) a0[r] = bb0;
    a0 = wmma16(A, Bt0, a0);
    v8f a1; for (int r = 0; r < 8; ++r) a1[r] = bb1;
    a1 = wmma16(A, Bt1, a1);
    for (int r = 0; r < 8; ++r) {
      hb[tile * 16 + r + half * 8][r16]      = a0[r];
      hb[tile * 16 + r + half * 8][16 + r16] = a1[r];
    }
  }
  __syncthreads();
  for (int o = tid; o < 512; o += 128) {
    const int pt = o >> 5, ch = o & 31;
    float m = -3.0e38f;
    for (int k = 0; k < KN; ++k) m = fmaxf(m, hb[pt * KN + k][ch]);
    x2[(size_t)(blockIdx.x * 16 + pt) * 32 + ch] = m;
  }
}

// ---------------- lin1 (48->128) + per-graph max pool; 8 waves, 1 N-tile each
__global__ __launch_bounds__(256) void lin1pool_kernel(
    const float* __restrict__ x1, const float* __restrict__ x2,
    const float* __restrict__ lw, const float* __restrict__ lb,
    float* __restrict__ g) {
  const int tid = threadIdx.x, lane = tid & 31, wave = tid >> 5;
  const int r16 = lane & 15, half = lane >> 4, kb = half * 8;
  const int ncol = wave * 16 + r16;

  v16h Bk0, Bk1;
  for (int j = 0; j < 16; ++j) {
    Bk0[j] = (_Float16)lw[(j < 8 ? kb + j : 8 + kb + j) * 128 + ncol];
    Bk1[j] = (j < 8) ? (_Float16)lw[(32 + kb + j) * 128 + ncol] : (_Float16)0;
  }
  const float bb = lb[ncol];

  float cmax = -3.0e38f;
  for (int m = 0; m < 8; ++m) {
    const int pt = blockIdx.x * PP + m * 16 + r16;
    const float* xi = x1 + (size_t)pt * 16;
    const float* xz = x2 + (size_t)pt * 32;
    v16h A0, A1;
    for (int j = 0; j < 8; ++j) {
      A0[j]     = (_Float16)xi[kb + j];        // ch 0..15
      A0[8 + j] = (_Float16)xz[kb + j];        // ch 16..31
      A1[j]     = (_Float16)xz[16 + kb + j];   // ch 32..47
      A1[8 + j] = (_Float16)0;                 // ch 48..63 pad
    }
    v8f acc; for (int r = 0; r < 8; ++r) acc[r] = bb;
    acc = wmma16(A0, Bk0, acc);
    acc = wmma16(A1, Bk1, acc);
    for (int r = 0; r < 8; ++r) cmax = fmaxf(cmax, acc[r]);
  }
  cmax = fmaxf(cmax, __shfl_xor(cmax, 16, 32));
  if (lane < 16) g[(size_t)blockIdx.x * 128 + ncol] = cmax;
}

// ------------------------------------- head MLP + log_softmax (per graph)
__global__ __launch_bounds__(64) void head_kernel(
    const float* __restrict__ g,
    const float* __restrict__ w1, const float* __restrict__ b1,
    const float* __restrict__ w2, const float* __restrict__ b2,
    const float* __restrict__ w3, const float* __restrict__ b3,
    float* __restrict__ out) {
  __shared__ float gb[128], h1[64], h2[64], ob[2];
  const int t = threadIdx.x, b = blockIdx.x;
  gb[t]      = g[(size_t)b * 128 + t];
  gb[t + 64] = g[(size_t)b * 128 + t + 64];
  __syncthreads();
  {
    float v = b1[t];
    for (int k = 0; k < 128; ++k) v += gb[k] * w1[k * 64 + t];
    h1[t] = fmaxf(v, 0.0f);
  }
  __syncthreads();
  {
    float v = b2[t];
    for (int k = 0; k < 64; ++k) v += h1[k] * w2[k * 64 + t];
    h2[t] = fmaxf(v, 0.0f);
  }
  __syncthreads();
  if (t < OUTC) {
    float v = b3[t];
    for (int k = 0; k < 64; ++k) v += h2[k] * w3[k * OUTC + t];
    ob[t] = v;
  }
  __syncthreads();
  if (t < OUTC) {
    float m = fmaxf(ob[0], ob[1]);
    float lse = m + logf(expf(ob[0] - m) + expf(ob[1] - m));
    out[(size_t)b * OUTC + t] = ob[t] - lse;
  }
}

// ----------------------------------------------------------------- launch
extern "C" void kernel_launch(void* const* d_in, const int* in_sizes, int n_in,
                              void* d_out, int out_size, void* d_ws, size_t ws_size,
                              hipStream_t stream) {
  const float* pos    = (const float*)d_in[0];
  // d_in[1] = batch (implicit layout, unused)
  const float* c1_w1  = (const float*)d_in[2];
  const float* c1_b1  = (const float*)d_in[3];
  const float* c1_g1  = (const float*)d_in[4];
  const float* c1_be1 = (const float*)d_in[5];
  const float* c1_w2  = (const float*)d_in[6];
  const float* c1_b2  = (const float*)d_in[7];
  const float* c1_g2  = (const float*)d_in[8];
  const float* c1_be2 = (const float*)d_in[9];
  const float* c1_w3  = (const float*)d_in[10];
  const float* c1_b3  = (const float*)d_in[11];
  const float* c2_w   = (const float*)d_in[12];
  const float* c2_b   = (const float*)d_in[13];
  const float* lw     = (const float*)d_in[14];
  const float* lb     = (const float*)d_in[15];
  const float* m_w1   = (const float*)d_in[16];
  const float* m_b1   = (const float*)d_in[17];
  const float* m_w2   = (const float*)d_in[18];
  const float* m_b2   = (const float*)d_in[19];
  const float* m_w3   = (const float*)d_in[20];
  const float* m_b3   = (const float*)d_in[21];
  float* out = (float*)d_out;

  // workspace layout (4-byte element offsets)
  int*   idx1 = (int*)d_ws;
  int*   idx2 = idx1 + (size_t)EE;
  float* ws   = (float*)d_ws;
  float* x1   = ws + 2 * (size_t)EE;
  float* x2   = x1 + (size_t)BP * 16;
  float* gbuf = x2 + (size_t)BP * 32;
  float* p1   = gbuf + (size_t)BB * 128;
  float* p2   = p1 + (size_t)STAT_BLOCKS * 32;
  float* bn1  = p2 + (size_t)STAT_BLOCKS * 32;
  float* bn2  = bn1 + 32;

  knn_wmma_kernel<3, 4><<<BB, 128, 0, stream>>>(pos, idx1);
  stats1_kernel<<<STAT_BLOCKS, 256, 0, stream>>>(pos, idx1, c1_w1, c1_b1, p1);
  bn_finalize_kernel<<<1, 16, 0, stream>>>(p1, STAT_BLOCKS, c1_g1, c1_be1, bn1);
  stats2_kernel<<<STAT_BLOCKS, 256, 0, stream>>>(pos, idx1, c1_w1, c1_b1, bn1,
                                                 c1_w2, c1_b2, p2);
  bn_finalize_kernel<<<1, 16, 0, stream>>>(p2, STAT_BLOCKS, c1_g2, c1_be2, bn2);
  conv1_kernel<<<BP / 16, 128, 0, stream>>>(pos, idx1, c1_w1, c1_b1, c1_w2, c1_b2,
                                            c1_w3, c1_b3, bn1, bn2, x1);
  knn_wmma_kernel<16, 16><<<BB, 128, 0, stream>>>(x1, idx2);
  conv2_kernel<<<BP / 16, 128, 0, stream>>>(x1, idx2, c2_w, c2_b, x2);
  lin1pool_kernel<<<BB, 256, 0, stream>>>(x1, x2, lw, lb, gbuf);
  head_kernel<<<BB, 64, 0, stream>>>(gbuf, m_w1, m_b1, m_w2, m_b2, m_w3, m_b3, out);
}
